// GINEncoder_30133490549166
// MI455X (gfx1250) — matrix operand order repaired and checked by reading, not compiled, not run on hardware
//
#include <hip/hip_runtime.h>

// ---------------------------------------------------------------------------
// GIN-style GNN encoder for MI455X (gfx1250, wave32, WMMA).
//   Layers: agg = scatter_add(h[src] -> dst); z = h + agg;
//           z = relu(z@W1+b1)@W2+b2; h = relu(z) (no relu last layer)
//   Output: per-graph mean pool over 'batch' (256 graphs x 128 feats).
// GEMMs use v_wmma_f32_16x16x32_f16 (f16 A/B, f32 accumulate), templated
// epilogue (no runtime branches), 2 row-tiles per block with B fragments
// register-resident across tiles.
// ---------------------------------------------------------------------------

typedef __attribute__((ext_vector_type(16))) _Float16 v16h;
typedef __attribute__((ext_vector_type(8)))  _Float16 v8h;
typedef __attribute__((ext_vector_type(4)))  _Float16 v4h;
typedef __attribute__((ext_vector_type(8)))  float    v8f;

#define N_NODES  100000
#define N_EDGES  1600000
#define NF       128
#define N_GRAPHS 256
#define N_LAYERS 5

// ---------------- zero fill (float4 granularity) ---------------------------
__global__ void k_zero4(float4* __restrict__ p, int n4) {
    int i = blockIdx.x * blockDim.x + threadIdx.x;
    if (i < n4) p[i] = make_float4(0.f, 0.f, 0.f, 0.f);
}

// ---------------- edge scatter-add: one wave per edge ----------------------
// lane l handles features [4l, 4l+4): float4 gather + 4 global f32 atomics.
__global__ void k_scatter(const float* __restrict__ h,
                          const int* __restrict__ src,
                          const int* __restrict__ dst,
                          float* __restrict__ agg, int n_edges) {
    int gid  = blockIdx.x * blockDim.x + threadIdx.x;
    int lane = gid & 31;
    int e    = gid >> 5;
    if (e >= n_edges) return;
    int s = src[e];
    int d = dst[e];
    const float4 v = *(const float4*)(h + (size_t)s * NF + lane * 4);
    float* ap = agg + (size_t)d * NF + lane * 4;
    atomicAdd(ap + 0, v.x);
    atomicAdd(ap + 1, v.y);
    atomicAdd(ap + 2, v.z);
    atomicAdd(ap + 3, v.w);
}

// ---------------- z = h + agg, converted to f16 ----------------------------
__global__ void k_addcvt(const float* __restrict__ h,
                         const float* __restrict__ agg,
                         _Float16* __restrict__ zh, int n4) {
    int i = blockIdx.x * blockDim.x + threadIdx.x;
    if (i >= n4) return;
    float4 a = ((const float4*)h)[i];
    float4 g = ((const float4*)agg)[i];
    v4h o;
    o.x = (_Float16)(a.x + g.x);
    o.y = (_Float16)(a.y + g.y);
    o.z = (_Float16)(a.z + g.z);
    o.w = (_Float16)(a.w + g.w);
    ((v4h*)zh)[i] = o;
}

// ---------------- pack W (128x128 f32 row-major) into WMMA B fragments -----
// Bp[((ks*8 + nt)*32 + lane)*16 + j]:
//   n = nt*16 + (lane&15),  k = ks*32 + (lane<16 ? 0 : 8) + (j&7) + (j&8 ? 16 : 0)
// (B 32x16 fragment: lanes 0-15 hold K 0..7,16..23, lanes 16-31 hold K 8..15,24..31 of
//  each 32-wide K step, matching the A-operand K interleave.)
__global__ void k_packw(const float* __restrict__ W, _Float16* __restrict__ Bp) {
    int t = blockIdx.x * blockDim.x + threadIdx.x;  // 0..16383
    if (t >= 4 * 8 * 32 * 16) return;
    int j    = t & 15;
    int lane = (t >> 4) & 31;
    int nt   = (t >> 9) & 7;
    int ks   = (t >> 12) & 3;
    int n = nt * 16 + (lane & 15);
    int k = ks * 32 + ((lane < 16) ? 0 : 8) + (j & 7) + ((j & 8) ? 16 : 0);
    Bp[t] = (_Float16)W[k * NF + n];
}

// ---------------- WMMA GEMM: [32 rows x 128] tile per block ----------------
// 256 threads = 8 waves; wave w owns output columns [16w, 16w+16) and both
// 16-row sub-tiles. B fragments live in registers across both sub-tiles.
// 8x v_wmma_f32_16x16x32_f16 per wave. EXEC is all-ones throughout (grid
// divides rows exactly; no divergence before the WMMAs).
template <bool OUTF16, bool RELU>
__global__ __launch_bounds__(256)
void k_gemm(const _Float16* __restrict__ A,   // [n_rows,128] f16 row-major
            const _Float16* __restrict__ Bp,  // packed fragments (16384 halves)
            const float* __restrict__ bias,   // [128]
            void* __restrict__ outp) {        // f16 or f32 [n_rows,128]
    __shared__ _Float16 Alds[32 * NF];        // 8 KB
    const int t    = threadIdx.x;
    const int row0 = blockIdx.x * 32;

    // cooperative A-tile load: 256 threads x 32 bytes
    {
        const int r   = t >> 3;   // 0..31
        const int sg  = t & 7;    // 0..7 (16 halves each)
        const size_t g = (size_t)(row0 + r) * NF + sg * 16;
        *(uint4*)(&Alds[r * NF + sg * 16])     = *(const uint4*)(&A[g]);
        *(uint4*)(&Alds[r * NF + sg * 16 + 8]) = *(const uint4*)(&A[g + 8]);
    }
    __syncthreads();

    const int wave = t >> 5;               // column tile 0..7
    const int lane = t & 31;
    const int m    = lane & 15;
    const int kb0  = (lane < 16) ? 0 : 8;  // ISA A-operand K interleave

    // B fragments: one 32B contiguous load per k-step, register-resident
    v16h b[4];
#pragma unroll
    for (int ks = 0; ks < 4; ++ks)
        b[ks] = *(const v16h*)(&Bp[(((ks * 8 + wave) * 32) + lane) * 16]);

    v8f c0 = {};
    v8f c1 = {};
#pragma unroll
    for (int ks = 0; ks < 4; ++ks) {
        // A fragment: elements 0..7 = K(ks*32+kb0+0..7), 8..15 = +16
        const _Float16* ap0 = &Alds[m * NF + ks * 32 + kb0];
        const _Float16* ap1 = ap0 + 16 * NF;
        v8h a0lo = *(const v8h*)ap0;
        v8h a0hi = *(const v8h*)(ap0 + 16);
        v8h a1lo = *(const v8h*)ap1;
        v8h a1hi = *(const v8h*)(ap1 + 16);
        v16h a0 = __builtin_shufflevector(a0lo, a0hi,
                                          0, 1, 2, 3, 4, 5, 6, 7,
                                          8, 9, 10, 11, 12, 13, 14, 15);
        v16h a1 = __builtin_shufflevector(a1lo, a1hi,
                                          0, 1, 2, 3, 4, 5, 6, 7,
                                          8, 9, 10, 11, 12, 13, 14, 15);
        c0 = __builtin_amdgcn_wmma_f32_16x16x32_f16(
            false, a0, false, b[ks], (short)0, c0, false, false);
        c1 = __builtin_amdgcn_wmma_f32_16x16x32_f16(
            false, a1, false, b[ks], (short)0, c1, false, false);
    }

    // C/D layout: lane<16 -> rows 0..7, lane>=16 -> rows 8..15; N = wave*16+m
    const int n     = wave * 16 + m;
    const int rbase = (lane < 16) ? 0 : 8;
    const float bv  = bias[n];
#pragma unroll
    for (int r = 0; r < 8; ++r) {
        float v0 = c0[r] + bv;
        float v1 = c1[r] + bv;
        if (RELU) { v0 = fmaxf(v0, 0.0f); v1 = fmaxf(v1, 0.0f); }
        const size_t i0 = (size_t)(row0 + rbase + r) * NF + n;
        const size_t i1 = i0 + (size_t)16 * NF;
        if (OUTF16) {
            ((_Float16*)outp)[i0] = (_Float16)v0;
            ((_Float16*)outp)[i1] = (_Float16)v1;
        } else {
            ((float*)outp)[i0] = v0;
            ((float*)outp)[i1] = v1;
        }
    }
}

// ---------------- graph mean pooling ---------------------------------------
__global__ void k_poolsum(const float* __restrict__ h,
                          const int* __restrict__ batch,
                          float* __restrict__ outsum,
                          float* __restrict__ cnt, int n_nodes) {
    int gid  = blockIdx.x * blockDim.x + threadIdx.x;
    int lane = gid & 31;
    int node = gid >> 5;
    if (node >= n_nodes) return;
    int g = batch[node];
    const float4 v = *(const float4*)(h + (size_t)node * NF + lane * 4);
    float* op = outsum + (size_t)g * NF + lane * 4;
    atomicAdd(op + 0, v.x);
    atomicAdd(op + 1, v.y);
    atomicAdd(op + 2, v.z);
    atomicAdd(op + 3, v.w);
    if (lane == 0) atomicAdd(cnt + g, 1.0f);
}

__global__ void k_pooldiv(const float* __restrict__ outsum,
                          const float* __restrict__ cnt,
                          float* __restrict__ out, int n) {
    int i = blockIdx.x * blockDim.x + threadIdx.x;
    if (i >= n) return;
    int g = i >> 7;  // /128
    out[i] = outsum[i] / fmaxf(cnt[g], 1.0f);
}

// ---------------------------------------------------------------------------
extern "C" void kernel_launch(void* const* d_in, const int* in_sizes, int n_in,
                              void* d_out, int out_size, void* d_ws, size_t ws_size,
                              hipStream_t stream) {
    const float* x     = (const float*)d_in[0];
    const int*   ei    = (const int*)d_in[1];
    const int*   batch = (const int*)d_in[2];
    const int*   src   = ei;
    const int*   dst   = ei + N_EDGES;

    // workspace carve-up (256B aligned)
    char* ws = (char*)d_ws;
    size_t off = 0;
    auto carve = [&](size_t bytes) {
        char* p = ws + off;
        off = (off + bytes + 255) & ~(size_t)255;
        return p;
    };
    float*    h      = (float*)   carve((size_t)N_NODES * NF * 4);   // 51.2 MB
    float*    agg    = (float*)   carve((size_t)N_NODES * NF * 4);   // 51.2 MB
    _Float16* zh16   = (_Float16*)carve((size_t)N_NODES * NF * 2);   // 25.6 MB
    _Float16* y16    = (_Float16*)carve((size_t)N_NODES * NF * 2);   // 25.6 MB
    _Float16* w1p    = (_Float16*)carve(16384 * 2);
    _Float16* w2p    = (_Float16*)carve(16384 * 2);
    float*    outsum = (float*)   carve((size_t)N_GRAPHS * NF * 4);
    float*    cnt    = (float*)   carve((size_t)N_GRAPHS * 4);

    const int feat4     = N_NODES * NF / 4;          // 3,200,000 float4s
    const int zeroBlks  = (feat4 + 255) / 256;
    const int scatBlks  = (N_EDGES * 32) / 256;      // 200,000
    const int gemmBlks  = N_NODES / 32;              // 3,125 (exact)
    const int poolBlks  = (N_NODES * 32 + 255) / 256;

    for (int layer = 0; layer < N_LAYERS; ++layer) {
        const float* hin = (layer == 0) ? x : h;
        const float* W1  = (const float*)d_in[3 + 4 * layer + 0];
        const float* b1  = (const float*)d_in[3 + 4 * layer + 1];
        const float* W2  = (const float*)d_in[3 + 4 * layer + 2];
        const float* b2  = (const float*)d_in[3 + 4 * layer + 3];

        k_zero4<<<zeroBlks, 256, 0, stream>>>((float4*)agg, feat4);
        k_packw<<<64, 256, 0, stream>>>(W1, w1p);
        k_packw<<<64, 256, 0, stream>>>(W2, w2p);
        k_scatter<<<scatBlks, 256, 0, stream>>>(hin, src, dst, agg, N_EDGES);
        k_addcvt<<<zeroBlks, 256, 0, stream>>>(hin, agg, zh16, feat4);
        // z' = relu(z@W1 + b1)  -> f16
        k_gemm<true, true><<<gemmBlks, 256, 0, stream>>>(zh16, w1p, b1, y16);
        // h  = (relu?)(z'@W2 + b2) -> f32 (overwrites h; stream-ordered, safe)
        if (layer < N_LAYERS - 1)
            k_gemm<false, true><<<gemmBlks, 256, 0, stream>>>(y16, w2p, b2, h);
        else
            k_gemm<false, false><<<gemmBlks, 256, 0, stream>>>(y16, w2p, b2, h);
    }

    // mean pool per graph
    k_zero4<<<(N_GRAPHS * NF / 4 + 255) / 256, 256, 0, stream>>>(
        (float4*)outsum, N_GRAPHS * NF / 4);
    k_zero4<<<1, 256, 0, stream>>>((float4*)cnt, N_GRAPHS / 4);
    k_poolsum<<<poolBlks, 256, 0, stream>>>(h, batch, outsum, cnt, N_NODES);
    k_pooldiv<<<(N_GRAPHS * NF + 255) / 256, 256, 0, stream>>>(
        outsum, cnt, (float*)d_out, N_GRAPHS * NF);

    (void)in_sizes; (void)n_in; (void)out_size; (void)ws_size;
}
